// GGUFWeight_30786325578068
// MI455X (gfx1250) — matrix operand order repaired and checked by reading, not compiled, not run on hardware
//
#include <hip/hip_runtime.h>

// ---- CDNA5 (gfx1250) vector types ----
typedef __attribute__((ext_vector_type(16))) __bf16 v16bf;
typedef __attribute__((ext_vector_type(8)))  __bf16 v8bf;
typedef __attribute__((ext_vector_type(8)))  float  v8f;

// Problem constants (from the reference)
constexpr int Bc = 4, Sc = 2048, Kc = 4096, Nc = 4096;
constexpr int Mc = Bc * Sc;            // 8192 rows of x (flattened [B*S, K])
constexpr int BLK = 32;                // Q4_0 block width along K

// Tiling: block = 128(M) x 256(N); 8 wave32s in a 2x4 grid, each wave 64x64
constexpr int TM = 128, TN = 256, TK = 32;
constexpr int LDSTR = 40;              // bf16 row stride in LDS (80 B, 16B-aligned)

// Workspace layout (bytes): xbf [Mc*Kc*2 = 64 MiB] | wbf [Nc*Kc*2 = 32 MiB]
constexpr size_t WS_XBF_OFF = 0;
constexpr size_t WS_WBF_OFF = (size_t)Mc * Kc * 2;

// ---------------- pre-pass 1: x fp32 -> bf16 ----------------
__global__ __launch_bounds__(256)
void cvt_x_bf16(const float* __restrict__ x, __bf16* __restrict__ xb)
{
    const size_t i = ((size_t)blockIdx.x * 256 + threadIdx.x) * 8;
    float4 a = *(const float4*)(x + i);
    float4 b = *(const float4*)(x + i + 4);
    v8bf o;
    o[0] = (__bf16)a.x; o[1] = (__bf16)a.y; o[2] = (__bf16)a.z; o[3] = (__bf16)a.w;
    o[4] = (__bf16)b.x; o[5] = (__bf16)b.y; o[6] = (__bf16)b.z; o[7] = (__bf16)b.w;
    *(v8bf*)(xb + i) = o;                       // global_store_b128
}

// ---------------- pre-pass 2: Q4_0 dequant -> bf16 ----------------
// w[n,k] = scale[n, k/32] * (q[n,k] - 8); one thread handles one 32-wide block
__global__ __launch_bounds__(256)
void deq_w_bf16(const int* __restrict__ qw, const float* __restrict__ sc,
                __bf16* __restrict__ wb)
{
    const size_t blk  = (size_t)blockIdx.x * 256 + threadIdx.x;  // Nc*Kc/32 blocks
    const size_t base = blk * BLK;
    const float  s    = sc[blk];
    #pragma unroll
    for (int j = 0; j < 4; ++j) {
        int4 qa = *(const int4*)(qw + base + j * 8);
        int4 qb = *(const int4*)(qw + base + j * 8 + 4);
        v8bf o;
        o[0] = (__bf16)((float)(qa.x - 8) * s);
        o[1] = (__bf16)((float)(qa.y - 8) * s);
        o[2] = (__bf16)((float)(qa.z - 8) * s);
        o[3] = (__bf16)((float)(qa.w - 8) * s);
        o[4] = (__bf16)((float)(qb.x - 8) * s);
        o[5] = (__bf16)((float)(qb.y - 8) * s);
        o[6] = (__bf16)((float)(qb.z - 8) * s);
        o[7] = (__bf16)((float)(qb.w - 8) * s);
        *(v8bf*)(wb + base + j * 8) = o;        // global_store_b128
    }
}

// ---------------- main GEMM: async LDS staging + WMMA ----------------
__device__ __forceinline__ void async_b128(unsigned lds_off, unsigned goff,
                                           const void* sbase)
{
    // VGLOBAL async copy: memory -> LDS, tracked by ASYNCcnt (no VGPR data path)
    asm volatile("global_load_async_to_lds_b128 %0, %1, %2"
                 :: "v"(lds_off), "v"(goff), "s"(sbase) : "memory");
}
__device__ __forceinline__ void wait_async0()
{
    asm volatile("s_wait_asynccnt 0" ::: "memory");
}

__global__ __launch_bounds__(256)
void q4gemm_wmma_bf16(const __bf16* __restrict__ xb,
                      const __bf16* __restrict__ wb,
                      const float*  __restrict__ bias,
                      float*        __restrict__ out)
{
    __shared__ __bf16 As[2][TM * LDSTR];        // x tile (bf16), ping-pong
    __shared__ __bf16 Bs[2][TN * LDSTR];        // weight tile (bf16), ping-pong

    const int tid  = threadIdx.x;
    const int m0   = blockIdx.x * TM;
    const int n0   = blockIdx.y * TN;
    const int lane = tid & 31;
    const int w    = tid >> 5;                  // wave id 0..7
    const int wm   = (w & 1) * 64;              // wave M offset (2 rows of waves)
    const int wn   = (w >> 1) * 64;             // wave N offset (4 cols of waves)
    const int h    = lane >> 4;
    const int ml   = lane & 15;

    v8f acc[4][4] = {};                         // 64x64 per wave: 16 accumulators

    // Issue one K-slab of async copies into buffer `buf`.
    // A tile: 128 rows x 64 B = 512 x 16B chunks (2/thread)
    // B tile: 256 rows x 64 B = 1024 x 16B chunks (4/thread)
    auto issue = [&](int buf, int ks) {
        const unsigned aBase = (unsigned)(size_t)(void*)&As[buf][0];
        const unsigned bBase = (unsigned)(size_t)(void*)&Bs[buf][0];
        #pragma unroll
        for (int t = 0; t < 2; ++t) {
            const int c   = tid + t * 256;
            const int row = c >> 2;
            const int q   = (c & 3) * 16;
            async_b128(aBase + (unsigned)row * (LDSTR * 2) + q,
                       ((unsigned)(m0 + row) * Kc + (unsigned)ks) * 2 + q, xb);
        }
        #pragma unroll
        for (int t = 0; t < 4; ++t) {
            const int c   = tid + t * 256;
            const int row = c >> 2;
            const int q   = (c & 3) * 16;
            async_b128(bBase + (unsigned)row * (LDSTR * 2) + q,
                       ((unsigned)(n0 + row) * Kc + (unsigned)ks) * 2 + q, wb);
        }
    };

    // Fragment loads (documented WMMA layouts) + 16 WMMAs from buffer `buf`.
    auto compute = [&](int buf) {
        const __bf16* A  = &As[buf][0];
        const __bf16* Bt = &Bs[buf][0];
        v16bf afrag[4];
        #pragma unroll
        for (int mi = 0; mi < 4; ++mi) {
            // A 16x32 bf16: lane 16h+m holds K in [8h,8h+8) then [16+8h,16+8h+8)
            const __bf16* p = A + (wm + mi * 16 + ml) * LDSTR + 8 * h;
            v8bf lo = *(const v8bf*)(p);
            v8bf hi = *(const v8bf*)(p + 16);
            afrag[mi] = __builtin_shufflevector(lo, hi,
                0,1,2,3,4,5,6,7,8,9,10,11,12,13,14,15);
        }
        v16bf bfrag[4];
        #pragma unroll
        for (int ni = 0; ni < 4; ++ni) {
            // B 32x16 bf16: lane 16h+n holds K in [8h,8h+8) then [16+8h,16+8h+8)
            const __bf16* p = Bt + (wn + ni * 16 + ml) * LDSTR + 8 * h;
            v8bf lo = *(const v8bf*)(p);
            v8bf hi = *(const v8bf*)(p + 16);
            bfrag[ni] = __builtin_shufflevector(lo, hi,
                0,1,2,3,4,5,6,7,8,9,10,11,12,13,14,15);
        }
        #pragma unroll
        for (int mi = 0; mi < 4; ++mi)
            #pragma unroll
            for (int ni = 0; ni < 4; ++ni)
                acc[mi][ni] = __builtin_amdgcn_wmma_f32_16x16x32_bf16(
                    false, afrag[mi], false, bfrag[ni],
                    (short)0, acc[mi][ni], false, false);
    };

    issue(0, 0);
    for (int ks = 0; ks < Kc; ks += 2 * TK) {
        // ---- stage 0 ----
        wait_async0();                           // buf0 slab resident in LDS
        __syncthreads();
        if (ks + TK < Kc) issue(1, ks + TK);     // overlap next slab with compute
        compute(0);
        // ---- stage 1 ----
        wait_async0();                           // buf1 slab resident
        __syncthreads();
        if (ks + 2 * TK < Kc) issue(0, ks + 2 * TK);
        compute(1);
    }

    // ---- epilogue: C/D layout (VGPR r -> row r+8h, lane%16 -> col), + bias ----
    #pragma unroll
    for (int mi = 0; mi < 4; ++mi) {
        #pragma unroll
        for (int ni = 0; ni < 4; ++ni) {
            const int n    = n0 + wn + ni * 16 + ml;
            const float bv = bias[n];
            #pragma unroll
            for (int r = 0; r < 8; ++r) {
                const int m = m0 + wm + mi * 16 + 8 * h + r;
                out[(size_t)m * Nc + n] = acc[mi][ni][r] + bv;
            }
        }
    }
}

extern "C" void kernel_launch(void* const* d_in, const int* in_sizes, int n_in,
                              void* d_out, int out_size, void* d_ws, size_t ws_size,
                              hipStream_t stream) {
    const float* x    = (const float*)d_in[0];  // [B,S,K] fp32
    const int*   qw   = (const int*)  d_in[1];  // [N,K] int32 codes 0..15
    const float* sc   = (const float*)d_in[2];  // [N,K/32] fp32
    const float* bias = (const float*)d_in[3];  // [N] fp32
    float*       out  = (float*)d_out;          // [B,S,N] fp32

    __bf16* xbf = (__bf16*)((char*)d_ws + WS_XBF_OFF);
    __bf16* wbf = (__bf16*)((char*)d_ws + WS_WBF_OFF);

    // pre-pass: fp32 x -> bf16 (8 elems/thread)
    cvt_x_bf16<<<dim3((Mc * (size_t)Kc) / (256 * 8)), dim3(256), 0, stream>>>(x, xbf);
    // pre-pass: Q4_0 -> bf16 (one 32-block/thread)
    deq_w_bf16<<<dim3((Nc * (size_t)(Kc / BLK)) / 256), dim3(256), 0, stream>>>(qw, sc, wbf);
    // main GEMM
    dim3 grid(Mc / TM, Nc / TN);                // 64 x 16 blocks
    q4gemm_wmma_bf16<<<grid, dim3(256), 0, stream>>>(xbf, wbf, bias, out);
}
